// MultiHeadAttention_35459249996110
// MI455X (gfx1250) — compile-verified, matrix operand order checked by reading, not run on hardware
//
#include <hip/hip_runtime.h>
#include <hip/hip_bf16.h>
#include <math.h>

typedef __bf16 bf16_t;
typedef __bf16 v16bf __attribute__((ext_vector_type(16)));
typedef float  v8f   __attribute__((ext_vector_type(8)));
typedef unsigned int v4u __attribute__((ext_vector_type(4)));
typedef int          v4i __attribute__((ext_vector_type(4)));
typedef int          v8i __attribute__((ext_vector_type(8)));

#define BATCH  4096
#define NBLK   32
#define DMODEL 512
#define HEADS  8
#define HDIM   64
#define TOPK   6
#define EPS    1e-7f

// ---------------------------------------------------------------------------
// LDS byte offset of a __shared__ object: generic LDS pointers on amdgcn carry
// the workgroup-relative offset in the low 32 bits (flat aperture mapping).
// ---------------------------------------------------------------------------
__device__ inline unsigned lds_addr_of(const void* p) {
  return (unsigned)(size_t)p;
}

// ---------------------------------------------------------------------------
// Tensor Data Mover: issue a 2D tile load Global -> LDS (ISA 08, D# groups).
//   tile_w      : tile dim0 (contiguous, in elements)
//   tile_h      : tile dim1 (rows)
//   row_stride  : tensor dim0 stride in elements
//   ds_log2     : data size encoding (0=1B,1=2B,2=4B,3=8B)
//   pad_int_enc : pad_interval encoding (2^(v+1) DWORDs before pad)
//   pad_amt_enc : pad_amount encoding (value+1 DWORDs of pad)
// Tensor dims are set equal to tile dims so no OOB clipping occurs.
// ---------------------------------------------------------------------------
__device__ inline void tdm_load_2d(unsigned lds_addr, const void* gptr,
                                   unsigned tile_w, unsigned tile_h,
                                   unsigned long long row_stride,
                                   unsigned ds_log2,
                                   unsigned pad_int_enc, unsigned pad_amt_enc,
                                   bool pad_en) {
  unsigned long long ga = (unsigned long long)gptr;
  v4u g0 = { 1u,                                   // count=1, load, not restore
             lds_addr,                             // LDS byte address
             (unsigned)(ga & 0xFFFFFFFFu),         // global_addr[31:0]
             (unsigned)((ga >> 32) & 0x1FFFFFFu)   // global_addr[56:32]
               | (2u << 30) };                     // type=2 ("image")
  unsigned w0 = (ds_log2 << 16)
              | (pad_en ? (1u << 20) : 0u)
              | (pad_int_enc << 22)
              | (pad_amt_enc << 25);
  v8i g1;
  g1[0] = (int)w0;                                         // mask=0, flags
  g1[1] = (int)((tile_w & 0xFFFFu) << 16);                 // tensor_dim0[15:0]
  g1[2] = (int)(((tile_w >> 16) & 0xFFFFu)                 // tensor_dim0[31:16]
              | ((tile_h & 0xFFFFu) << 16));               // tensor_dim1[15:0]
  g1[3] = (int)(((tile_h >> 16) & 0xFFFFu)                 // tensor_dim1[31:16]
              | ((tile_w & 0xFFFFu) << 16));               // tile_dim0
  g1[4] = (int)(tile_h & 0xFFFFu);                         // tile_dim1 (tile_dim2=0)
  g1[5] = (int)(row_stride & 0xFFFFFFFFull);               // dim0_stride[31:0]
  g1[6] = (int)((row_stride >> 32) & 0xFFFFull);           // dim0_stride[47:32]
  g1[7] = 0;
  v4i gz4 = { 0, 0, 0, 0 };
  v8i gz8 = { 0, 0, 0, 0, 0, 0, 0, 0 };
  // amdgpu-toolchain (clang-23) 6-arg form
  __builtin_amdgcn_tensor_load_to_lds(g0, g1, gz4, gz4, gz8, 0);
}

// ---------------------------------------------------------------------------
// WMMA fragment loaders. Per-lane layout for v_wmma_f32_16x16x32_bf16 (wave32):
//   lane & 15 -> row (M or N), (lane>>4)*8 -> K phase,
//   elems 0..7 = K[phase..phase+7], elems 8..15 = K[phase+16..phase+23]
// ---------------------------------------------------------------------------
__device__ inline v16bf load_frag(const bf16_t* lds, int rowoff, int stride) {
  int lane = threadIdx.x & 31;
  int r    = lane & 15;
  int kb   = (lane >> 4) * 8;
  const bf16_t* p = lds + (rowoff + r) * stride + kb;
  v16bf f;
#pragma unroll
  for (int t = 0; t < 8; ++t) { f[t] = p[t]; f[t + 8] = p[t + 16]; }
  return f;
}

__device__ inline v16bf load_frag_f32(const float* lds, int rowoff, int stride) {
  int lane = threadIdx.x & 31;
  int r    = lane & 15;
  int kb   = (lane >> 4) * 8;
  const float* p = lds + (rowoff + r) * stride + kb;
  v16bf f;
#pragma unroll
  for (int t = 0; t < 8; ++t) { f[t] = (bf16_t)p[t]; f[t + 8] = (bf16_t)p[t + 16]; }
  return f;
}

// ---------------------------------------------------------------------------
// Group (block-wise) GEMM: C[m, n, e] = sum_k A[m, n, k] * W[n, k, e]
// A fp32 [BATCH, NBLK, DMODEL] (TDM-staged fp32 tile, cvt at frag load),
// W fp32 (manual transpose+cvt to LDS), C bf16.
// Grid: (BATCH/64, DMODEL/64, NBLK); 128 threads = 4 waves; each wave 32x32.
// ---------------------------------------------------------------------------
__global__ __launch_bounds__(128)
void group_gemm_kernel(const float* __restrict__ A, const float* __restrict__ W,
                       bf16_t* __restrict__ C) {
  const int LDSA = 36;  // fp32 elems: 32 + 4-DWORD TDM pad
  const int LDSB = 40;  // bf16 elems
  __shared__ float  As[64 * LDSA];
  __shared__ bf16_t Bt[64 * LDSB];

  const int mbase = blockIdx.x * 64;
  const int nbase = blockIdx.y * 64;
  const int nb    = blockIdx.z;

  const int tid  = threadIdx.x;
  const int wave = tid >> 5;
  const int wm   = (wave >> 1) * 32;
  const int wn   = (wave & 1) * 32;

  v8f acc[2][2];
#pragma unroll
  for (int i = 0; i < 2; ++i)
#pragma unroll
    for (int j = 0; j < 2; ++j)
#pragma unroll
      for (int t = 0; t < 8; ++t) acc[i][j][t] = 0.f;

  const unsigned asA = lds_addr_of(As);

  for (int kt = 0; kt < DMODEL; kt += 32) {
    // TDM: 64x32 fp32 tile, rows strided by NBLK*DMODEL, pad 128B->+16B (stride 36)
    if (wave == 0) {
      const float* gA = A + ((size_t)mbase * NBLK + nb) * DMODEL + kt;
      tdm_load_2d(asA, gA, /*w*/32, /*h*/64,
                  (unsigned long long)NBLK * DMODEL,
                  /*ds_log2*/2, /*pad_int*/4, /*pad_amt*/3, true);
    }
    // stage W tile transposed (fp32 -> bf16); TDM cannot transpose
    {
      int kk = tid >> 2, qq = tid & 3;
      const float* g = W + ((size_t)nb * DMODEL + kt + kk) * DMODEL + nbase + qq * 16;
#pragma unroll
      for (int i = 0; i < 16; ++i) Bt[(qq * 16 + i) * LDSB + kk] = (bf16_t)g[i];
    }
    __builtin_amdgcn_s_wait_tensorcnt(0);
    __syncthreads();

#pragma unroll
    for (int i = 0; i < 2; ++i) {
      v16bf a = load_frag_f32(As, wm + i * 16, LDSA);
#pragma unroll
      for (int j = 0; j < 2; ++j) {
        v16bf b = load_frag(Bt, wn + j * 16, LDSB);
        acc[i][j] = __builtin_amdgcn_wmma_f32_16x16x32_bf16(
            false, a, false, b, (short)0, acc[i][j], false, false);
      }
    }
    __syncthreads();
  }

  const int lane = tid & 31;
  const int rlo  = (lane >> 4) * 8;
  const int col  = lane & 15;
#pragma unroll
  for (int i = 0; i < 2; ++i)
#pragma unroll
    for (int j = 0; j < 2; ++j)
#pragma unroll
      for (int v = 0; v < 8; ++v) {
        int m = mbase + wm + i * 16 + rlo + v;
        int e = nbase + wn + j * 16 + col;
        C[((size_t)m * NBLK + nb) * DMODEL + e] = (bf16_t)acc[i][j][v];
      }
}

// ---------------------------------------------------------------------------
// Attention core: one wave per (b, h). K/V head tiles TDM-staged into LDS as
// bf16. Scores -> softmax -> top-6 sparsify -> weighted sum over V.
// ---------------------------------------------------------------------------
__global__ __launch_bounds__(32)
void attention_kernel(const bf16_t* __restrict__ qh, const bf16_t* __restrict__ kh,
                      const bf16_t* __restrict__ vh, bf16_t* __restrict__ outb,
                      float* __restrict__ attn_out) {
  __shared__ bf16_t ks[32 * HDIM];
  __shared__ bf16_t vs[32 * HDIM];

  const int bh   = blockIdx.x;
  const int b    = bh >> 3;      // HEADS == 8
  const int h    = bh & 7;
  const int lane = threadIdx.x;  // 0..31, one q-row per lane

  // TDM: two 32x64 bf16 tiles (row stride DMODEL), no padding
  {
    size_t base = (size_t)(b * 32) * DMODEL + h * HDIM;
    tdm_load_2d(lds_addr_of(ks), kh + base, /*w*/HDIM, /*h*/32,
                (unsigned long long)DMODEL, /*ds_log2*/1, 0, 0, false);
    tdm_load_2d(lds_addr_of(vs), vh + base, /*w*/HDIM, /*h*/32,
                (unsigned long long)DMODEL, /*ds_log2*/1, 0, 0, false);
  }

  // q row in registers while TDM flies
  float qr[HDIM];
  size_t qbase = ((size_t)(b * 32 + lane)) * DMODEL + h * HDIM;
#pragma unroll
  for (int d = 0; d < HDIM; ++d) qr[d] = (float)qh[qbase + d];

  __builtin_amdgcn_s_wait_tensorcnt(0);  // single-wave workgroup: no barrier needed

  // scores, scale = 1/sqrt(64)
  float p[32];
#pragma unroll
  for (int k = 0; k < 32; ++k) {
    float s = 0.f;
#pragma unroll
    for (int d = 0; d < HDIM; ++d) s += qr[d] * (float)ks[k * HDIM + d];
    p[k] = s * 0.125f;
  }

  // softmax
  float mx = -3.4e38f;
#pragma unroll
  for (int k = 0; k < 32; ++k) mx = fmaxf(mx, p[k]);
  float sum = 0.f;
#pragma unroll
  for (int k = 0; k < 32; ++k) { p[k] = __expf(p[k] - mx); sum += p[k]; }
  float inv = 1.f / sum;
#pragma unroll
  for (int k = 0; k < 32; ++k) p[k] *= inv;

  // top-6 threshold (6th largest, duplicates counted) via masked max passes
  unsigned used = 0u;
  float th = 0.f;
#pragma unroll
  for (int it = 0; it < TOPK; ++it) {
    float best = -1.f;
    int   bi   = 0;
#pragma unroll
    for (int k = 0; k < 32; ++k) {
      bool free_k = ((used >> k) & 1u) == 0u;
      if (free_k && p[k] > best) { best = p[k]; bi = k; }
    }
    used |= (1u << bi);
    th = best;
  }
  th += EPS;

  float wsum = 0.f;
#pragma unroll
  for (int k = 0; k < 32; ++k) { float w = fmaxf(p[k] - th, 0.f); p[k] = w; wsum += w; }
  float winv = 1.f / (wsum + EPS);
#pragma unroll
  for (int k = 0; k < 32; ++k) p[k] *= winv;

  // attention weights out: [H*B, 32, 32] (torch layout)
  float* ap = attn_out + (((size_t)h * BATCH + b) * 32 + lane) * 32;
#pragma unroll
  for (int k = 0; k < 32; ++k) ap[k] = p[k];

  // context = attn @ V
  float acc[HDIM];
#pragma unroll
  for (int d = 0; d < HDIM; ++d) acc[d] = 0.f;
  for (int k = 0; k < 32; ++k) {
    float w = p[k];
#pragma unroll
    for (int d = 0; d < HDIM; ++d) acc[d] += w * (float)vs[k * HDIM + d];
  }
  size_t obase = ((size_t)(b * 32 + lane)) * DMODEL + h * HDIM;
#pragma unroll
  for (int d = 0; d < HDIM; ++d) outb[obase + d] = (bf16_t)acc[d];
}

// ---------------------------------------------------------------------------
// Fused fc + gate GEMMs with gating epilogue:
//   out = sigmoid(A@Wg + bg) * tanh(A@Wf + bf)
// A bf16 (TDM-staged with pad), Wf/Wg fp32 (manual transpose+cvt), out fp32.
// ---------------------------------------------------------------------------
__global__ __launch_bounds__(128)
void fc_gate_kernel(const bf16_t* __restrict__ A,
                    const float* __restrict__ Wf, const float* __restrict__ bf,
                    const float* __restrict__ Wg, const float* __restrict__ bg,
                    float* __restrict__ out) {
  const int LDST = 40;  // bf16 elems: 32 + 8-bf16 TDM pad
  __shared__ bf16_t As[64 * LDST];
  __shared__ bf16_t Ft[64 * LDST];
  __shared__ bf16_t Gt[64 * LDST];

  const int mbase = blockIdx.x * 64;
  const int nbase = blockIdx.y * 64;

  const int tid  = threadIdx.x;
  const int wave = tid >> 5;
  const int wm   = (wave >> 1) * 32;
  const int wn   = (wave & 1) * 32;

  v8f accF[2][2], accG[2][2];
#pragma unroll
  for (int i = 0; i < 2; ++i)
#pragma unroll
    for (int j = 0; j < 2; ++j)
#pragma unroll
      for (int t = 0; t < 8; ++t) { accF[i][j][t] = 0.f; accG[i][j][t] = 0.f; }

  const unsigned asA = lds_addr_of(As);

  for (int kt = 0; kt < DMODEL; kt += 32) {
    // TDM: 64x32 bf16 tile, pad 64B->+16B (stride 40 bf16)
    if (wave == 0) {
      const bf16_t* gA = A + (size_t)mbase * DMODEL + kt;
      tdm_load_2d(asA, gA, /*w*/32, /*h*/64, (unsigned long long)DMODEL,
                  /*ds_log2*/1, /*pad_int*/3, /*pad_amt*/3, true);
    }
    // stage both weight tiles transposed (fp32 -> bf16)
    {
      int kk = tid >> 2, qq = tid & 3;
      size_t goff = (size_t)(kt + kk) * DMODEL + nbase + qq * 16;
#pragma unroll
      for (int i = 0; i < 16; ++i) {
        Ft[(qq * 16 + i) * LDST + kk] = (bf16_t)Wf[goff + i];
        Gt[(qq * 16 + i) * LDST + kk] = (bf16_t)Wg[goff + i];
      }
    }
    __builtin_amdgcn_s_wait_tensorcnt(0);
    __syncthreads();

#pragma unroll
    for (int i = 0; i < 2; ++i) {
      v16bf a = load_frag(As, wm + i * 16, LDST);
#pragma unroll
      for (int j = 0; j < 2; ++j) {
        v16bf f = load_frag(Ft, wn + j * 16, LDST);
        accF[i][j] = __builtin_amdgcn_wmma_f32_16x16x32_bf16(
            false, a, false, f, (short)0, accF[i][j], false, false);
        v16bf g = load_frag(Gt, wn + j * 16, LDST);
        accG[i][j] = __builtin_amdgcn_wmma_f32_16x16x32_bf16(
            false, a, false, g, (short)0, accG[i][j], false, false);
      }
    }
    __syncthreads();
  }

  const int lane = tid & 31;
  const int rlo  = (lane >> 4) * 8;
  const int col  = lane & 15;
#pragma unroll
  for (int i = 0; i < 2; ++i)
#pragma unroll
    for (int j = 0; j < 2; ++j)
#pragma unroll
      for (int v = 0; v < 8; ++v) {
        int m = mbase + wm + i * 16 + rlo + v;
        int e = nbase + wn + j * 16 + col;
        float fcv = accF[i][j][v] + bf[e];
        float gv  = accG[i][j][v] + bg[e];
        float gate = 1.f / (1.f + __expf(-gv));
        out[(size_t)m * DMODEL + e] = gate * tanhf(fcv);
      }
}

// ---------------------------------------------------------------------------
extern "C" void kernel_launch(void* const* d_in, const int* in_sizes, int n_in,
                              void* d_out, int out_size, void* d_ws, size_t ws_size,
                              hipStream_t stream) {
  const float* q      = (const float*)d_in[0];
  const float* k      = (const float*)d_in[1];
  const float* v      = (const float*)d_in[2];
  const float* Wq     = (const float*)d_in[3];
  const float* Wk     = (const float*)d_in[4];
  const float* Wv     = (const float*)d_in[5];
  const float* fc_w   = (const float*)d_in[6];
  const float* fc_b   = (const float*)d_in[7];
  const float* gate_w = (const float*)d_in[8];
  const float* gate_b = (const float*)d_in[9];

  float* out      = (float*)d_out;
  float* attn_out = out + (size_t)BATCH * NBLK * DMODEL;  // second tuple element

  const size_t S = (size_t)BATCH * NBLK * DMODEL;  // 67,108,864 elems
  bf16_t* qh = (bf16_t*)d_ws;
  bf16_t* kh = qh + S;
  bf16_t* vh = kh + S;
  bf16_t* ob = vh + S;  // attention context, bf16

  dim3 gproj(BATCH / 64, DMODEL / 64, NBLK);
  group_gemm_kernel<<<gproj, 128, 0, stream>>>(q, Wq, qh);
  group_gemm_kernel<<<gproj, 128, 0, stream>>>(k, Wk, kh);
  group_gemm_kernel<<<gproj, 128, 0, stream>>>(v, Wv, vh);

  attention_kernel<<<BATCH * HEADS, 32, 0, stream>>>(qh, kh, vh, ob, attn_out);

  dim3 gfc((BATCH * NBLK) / 64, DMODEL / 64, 1);
  fc_gate_kernel<<<gfc, 128, 0, stream>>>(ob, fc_w, fc_b, gate_w, gate_b, out);
}
